// SlidingWindowAttention_63196148793901
// MI455X (gfx1250) — matrix operand order repaired and checked by reading, not compiled, not run on hardware
//
#include <hip/hip_runtime.h>
#include <hip/hip_bf16.h>

typedef __attribute__((ext_vector_type(16))) __bf16 v16bf;
typedef __attribute__((ext_vector_type(8)))  float v8f;

// ---------- helpers ----------

__device__ __forceinline__ unsigned short f2bf(float f) {
    unsigned u = __float_as_uint(f);
    u += 0x7FFFu + ((u >> 16) & 1u);          // round-to-nearest-even
    return (unsigned short)(u >> 16);
}

// A-matrix fragment (16x32 bf16, M in lane%16, K packed per ISA table):
// lanes 0-15: VGPR0..3 -> K 0..7, VGPR4..7 -> K 16..23 ; lanes 16-31: +8
__device__ __forceinline__ v16bf ld_a_frag(const unsigned short* p /*&T[m][kbase]*/, int hi) {
    union { v16bf v; unsigned u[8]; } f;
    const unsigned* q = (const unsigned*)p;   // 2 bf16 per dword
    int h = hi ? 4 : 0;
#pragma unroll
    for (int pp = 0; pp < 8; ++pp)
        f.u[pp] = q[((pp & 4) ? 8 : 0) + h + (pp & 3)];
    return f.v;
}

// B-matrix fragment (32x16 bf16, N in lane%16, K per ISA table): lanes 0-15 K 0..7,16..23,
// lanes 16-31 K 8..15,24..31. T is stored [n][k].
__device__ __forceinline__ v16bf ld_b_frag(const unsigned short* p /*&T[n][kbase]*/, int hi) {
    union { v16bf v; unsigned u[8]; } f;
    const unsigned* q = (const unsigned*)p;
    int h = hi ? 4 : 0;
#pragma unroll
    for (int pp = 0; pp < 8; ++pp)
        f.u[pp] = q[((pp & 4) ? 8 : 0) + h + (pp & 3)];
    return f.v;
}

__device__ __forceinline__ v8f wmma_bf16(v16bf a, v16bf b, v8f c) {
    return __builtin_amdgcn_wmma_f32_16x16x32_bf16(false, a, false, b, (short)0, c, false, false);
}

// CDNA5 async global->LDS copy (16B per lane), tracked with ASYNCcnt.
__device__ __forceinline__ unsigned lds_off(const void* p) {
    return (unsigned)(unsigned long long)(uintptr_t)p;   // low 32 bits = LDS byte offset
}
__device__ __forceinline__ void async_b128(const void* gaddr, unsigned ldsaddr) {
    asm volatile("global_load_async_to_lds_b128 %0, %1, off"
                 :: "v"(ldsaddr), "v"((unsigned long long)(uintptr_t)gaddr)
                 : "memory");
}
__device__ __forceinline__ void wait_async0() {
    asm volatile("s_wait_asynccnt 0" ::: "memory");
}

// ---------- GEMM: out[M][N] = X[M][K] @ W[K][N] (W always f32) ----------
// 256 threads = 8 waves; block tile 128x128; wave tile 32x64 (2x4 WMMA);
// register double-buffering of the global tile hides load latency under WMMA.

template <bool XBF16, bool OUTBF16>
__global__ __launch_bounds__(256) void gemm128(const void* __restrict__ Xv,
                                               const float* __restrict__ W,
                                               void* __restrict__ out,
                                               int M, int N, int K, float scale) {
    __shared__ unsigned short Xs[128][36];    // [m][k], 32-wide K tile, pad->36
    __shared__ unsigned short WsT[128][36];   // [n][k]
    const int tid  = threadIdx.x;
    const int lane = tid & 31, wv = tid >> 5;
    const int hi = (lane >> 4) & 1, l4 = lane & 15;
    const int m0 = blockIdx.x * 128, n0 = blockIdx.y * 128;
    const int wr = (wv >> 1) * 32;   // 0,32,64,96
    const int wc = (wv & 1) * 64;    // 0,64

    unsigned short rX[16], rW[16];
    auto fetch = [&](int kk) {
#pragma unroll
        for (int i = 0; i < 16; ++i) {
            int idx = tid + i * 256;                 // 4096 elems each
            int r = idx >> 5, c = idx & 31;
            if (XBF16) rX[i] = ((const unsigned short*)Xv)[(size_t)(m0 + r) * K + kk + c];
            else       rX[i] = f2bf(((const float*)Xv)[(size_t)(m0 + r) * K + kk + c]);
            int n = idx & 127, kq = idx >> 7;
            rW[i] = f2bf(W[(size_t)(kk + kq) * N + n0 + n]);
        }
    };

    v8f acc[2][4] = {};
    fetch(0);
    for (int kk = 0; kk < K; kk += 32) {
#pragma unroll
        for (int i = 0; i < 16; ++i) {
            int idx = tid + i * 256;
            Xs[idx >> 5][idx & 31] = rX[i];
            WsT[idx & 127][idx >> 7] = rW[i];
        }
        __syncthreads();
        if (kk + 32 < K) fetch(kk + 32);   // prefetch next tile during WMMAs

        v16bf a[2], b[4];
#pragma unroll
        for (int i = 0; i < 2; ++i) a[i] = ld_a_frag(&Xs[wr + i * 16 + l4][0], hi);
#pragma unroll
        for (int j = 0; j < 4; ++j) b[j] = ld_b_frag(&WsT[wc + j * 16 + l4][0], hi);
#pragma unroll
        for (int i = 0; i < 2; ++i)
#pragma unroll
            for (int j = 0; j < 4; ++j)
                acc[i][j] = wmma_bf16(a[i], b[j], acc[i][j]);
        __syncthreads();
    }

#pragma unroll
    for (int i = 0; i < 2; ++i)
#pragma unroll
        for (int j = 0; j < 4; ++j)
#pragma unroll
            for (int r = 0; r < 8; ++r) {
                int row = m0 + wr + i * 16 + r + hi * 8;
                int col = n0 + wc + j * 16 + l4;
                float val = acc[i][j][r] * scale;
                if (OUTBF16) ((unsigned short*)out)[(size_t)row * N + col] = f2bf(val);
                else         ((float*)out)[(size_t)row * N + col] = val;
            }
}

// ---------- fused attention (flash-style streaming softmax) ----------
// mode 0: local window (12 masked tiles) + 64 global keys, per (chunk,qtile,head)
// mode 1: global attention for rows [0,64) over all keys, per head
// q is pre-scaled by HD^-0.5. q/k/v/attn are bf16 [S][H*HD].

__global__ __launch_bounds__(128) void attn_fused(const unsigned short* __restrict__ q,
                                                  const unsigned short* __restrict__ k,
                                                  const unsigned short* __restrict__ v,
                                                  unsigned short* __restrict__ attn,
                                                  int mode) {
    const int S = 4096, Dm = 768, HDh = 64, Cc = 256;
    __shared__ unsigned short Qs[64][72];
    __shared__ unsigned short Ks[64][72];     // [key][d]
    __shared__ unsigned short VsT[64][72];    // [d][key]
    __shared__ unsigned short Ps[4][16][72];  // per-wave P transpose scratch

    const int tid = threadIdx.x, lane = tid & 31, wv = tid >> 5;
    const int hi = (lane >> 4) & 1, l4 = lane & 15;
    const int head = blockIdx.y;

    int qb, ntiles, winstart;
    if (mode == 0) {
        int chunk = blockIdx.x >> 2, qt = blockIdx.x & 3;
        qb = chunk * Cc + qt * 64;
        ntiles = 13;                 // 12 window tiles + 1 global-key tile
        winstart = chunk * Cc - Cc;
    } else {
        qb = 0; ntiles = S / 64; winstart = 0;
    }

    // stage Q (64 rows x 64 d) via async copy: 512 16B chunks, 4 per thread
#pragma unroll
    for (int i = 0; i < 4; ++i) {
        int idx = tid + i * 128;
        int r = idx >> 3, c = idx & 7;
        async_b128(q + (size_t)(qb + r) * Dm + head * HDh + c * 8,
                   lds_off(&Qs[r][c * 8]));
    }
    wait_async0();
    __syncthreads();

    v16bf aq0 = ld_a_frag(&Qs[wv * 16 + l4][0], hi);
    v16bf aq1 = ld_a_frag(&Qs[wv * 16 + l4][32], hi);

    v8f o[4] = {};
    float mrow[8], lrow[8];
#pragma unroll
    for (int r = 0; r < 8; ++r) { mrow[r] = -3.0e38f; lrow[r] = 0.f; }

    for (int t = 0; t < ntiles; ++t) {
        int pos0; bool masked;
        if (mode == 0) {
            if (t < 12) { pos0 = winstart + t * 64; masked = true; }
            else        { pos0 = 0;                 masked = false; }   // global keys
        } else { pos0 = t * 64; masked = false; }

        __syncthreads();
        // stage K [key][d] async (row-contiguous copy), clamped like reference's pc
#pragma unroll
        for (int i = 0; i < 4; ++i) {
            int idx = tid + i * 128;
            int r = idx >> 3, c = idx & 7;
            int pos = pos0 + r;
            int pc = pos < 0 ? 0 : (pos >= S ? S - 1 : pos);
            async_b128(k + (size_t)pc * Dm + head * HDh + c * 8,
                       lds_off(&Ks[r][c * 8]));
        }
        // stage V^T [d][key] via normal loads (transpose during store)
        for (int i = 0; i < 32; ++i) {
            int idx = tid + i * 128;
            int d = idx >> 6, r = idx & 63;
            int pos = pos0 + r;
            int pc = pos < 0 ? 0 : (pos >= S ? S - 1 : pos);
            VsT[d][r] = v[(size_t)pc * Dm + head * HDh + d];
        }
        wait_async0();
        __syncthreads();

        // scores: S(16q x 64k) as four 16x16 WMMA accumulators
        v8f s[4];
#pragma unroll
        for (int st = 0; st < 4; ++st) {
            v8f a = {};
            a = wmma_bf16(aq0, ld_b_frag(&Ks[st * 16 + l4][0], hi), a);
            a = wmma_bf16(aq1, ld_b_frag(&Ks[st * 16 + l4][32], hi), a);
            s[st] = a;
        }
        if (masked) {
#pragma unroll
            for (int st = 0; st < 4; ++st) {
                int pos = pos0 + st * 16 + l4;
                if (pos < 0 || pos >= S) {
#pragma unroll
                    for (int r = 0; r < 8; ++r) s[st][r] = -1.0e30f;
                }
            }
        }

        // flash softmax update (row = r + 8*hi; reduce across 16-lane group)
        float alpha[8], pv[4][8];
#pragma unroll
        for (int r = 0; r < 8; ++r) {
            float mx = fmaxf(fmaxf(s[0][r], s[1][r]), fmaxf(s[2][r], s[3][r]));
#pragma unroll
            for (int sh = 1; sh < 16; sh <<= 1) mx = fmaxf(mx, __shfl_xor(mx, sh, 32));
            float mnew = fmaxf(mrow[r], mx);
            alpha[r] = __expf(mrow[r] - mnew);
            mrow[r] = mnew;
            float psum = 0.f;
#pragma unroll
            for (int st = 0; st < 4; ++st) {
                float p = __expf(s[st][r] - mnew);
                pv[st][r] = p; psum += p;
            }
#pragma unroll
            for (int sh = 1; sh < 16; sh <<= 1) psum += __shfl_xor(psum, sh, 32);
            lrow[r] = lrow[r] * alpha[r] + psum;
        }
#pragma unroll
        for (int dt = 0; dt < 4; ++dt)
#pragma unroll
            for (int r = 0; r < 8; ++r) o[dt][r] *= alpha[r];

        // P (C-layout) -> LDS -> A-layout fragments (same-wave DS ops are in-order)
#pragma unroll
        for (int st = 0; st < 4; ++st)
#pragma unroll
            for (int r = 0; r < 8; ++r)
                Ps[wv][r + hi * 8][st * 16 + l4] = f2bf(pv[st][r]);

        v16bf ap0 = ld_a_frag(&Ps[wv][l4][0], hi);
        v16bf ap1 = ld_a_frag(&Ps[wv][l4][32], hi);
#pragma unroll
        for (int dt = 0; dt < 4; ++dt) {
            o[dt] = wmma_bf16(ap0, ld_b_frag(&VsT[dt * 16 + l4][0], hi), o[dt]);
            o[dt] = wmma_bf16(ap1, ld_b_frag(&VsT[dt * 16 + l4][32], hi), o[dt]);
        }
    }

    // finalize: attn[row][head*64 + d] = o / l
#pragma unroll
    for (int dt = 0; dt < 4; ++dt)
#pragma unroll
        for (int r = 0; r < 8; ++r) {
            int row = qb + wv * 16 + r + hi * 8;
            int col = head * HDh + dt * 16 + l4;
            attn[(size_t)row * Dm + col] = f2bf(o[dt][r] / lrow[r]);
        }
}

// ---------- launch ----------

extern "C" void kernel_launch(void* const* d_in, const int* in_sizes, int n_in,
                              void* d_out, int out_size, void* d_ws, size_t ws_size,
                              hipStream_t stream) {
    (void)in_sizes; (void)n_in; (void)out_size; (void)ws_size;
    const float* X  = (const float*)d_in[0];
    const float* Wq = (const float*)d_in[1];
    const float* Wk = (const float*)d_in[2];
    const float* Wv = (const float*)d_in[3];
    const float* Wo = (const float*)d_in[4];
    float* out = (float*)d_out;

    const int S = 4096, D = 768;
    const size_t SD = (size_t)S * D;
    unsigned short* qb = (unsigned short*)d_ws;
    unsigned short* kb = qb + SD;
    unsigned short* vb = kb + SD;
    unsigned short* ab = vb + SD;

    dim3 grid(S / 128, D / 128), block(256);

    // projections (scale = HD^-0.5 folded into q)
    gemm128<false, true><<<grid, block, 0, stream>>>(X, Wq, qb, S, D, D, 0.125f);
    gemm128<false, true><<<grid, block, 0, stream>>>(X, Wk, kb, S, D, D, 1.0f);
    gemm128<false, true><<<grid, block, 0, stream>>>(X, Wv, vb, S, D, D, 1.0f);

    // local windowed attention (all rows), then overwrite rows [0,64) with global attention
    attn_fused<<<dim3(64, 12), dim3(128), 0, stream>>>(qb, kb, vb, ab, 0);
    attn_fused<<<dim3(1, 12), dim3(128), 0, stream>>>(qb, kb, vb, ab, 1);

    // output projection
    gemm128<true, false><<<grid, block, 0, stream>>>(ab, Wo, out, S, D, D, 1.0f);
}
